// TwoDimEquivalent_10977936409119
// MI455X (gfx1250) — compile-verified, matrix operand order checked by reading, not run on hardware
//
#include <hip/hip_runtime.h>
#include <stdint.h>

typedef __attribute__((ext_vector_type(16))) _Float16 v16h;
typedef __attribute__((ext_vector_type(8)))  float    v8f;

#define NQ      64
#define ABND    5.0
#define NTAB    4096
#define DMAX    16.0f
#define CHUNK   128
#define RSTEP   0.2f              /* DT/TAU = 20/100 */
#define INV_2PI 0.15915494309189535f

// ---------------------------------------------------------------------------
// Kernel 1: 64-point Gauss-Legendre nodes/weights on [-1,1], scaled to [-A,A]
// with the exp(-z^2/2) factor folded into the weights (matches reference).
// Newton iteration on the Legendre recurrence, fp64, one thread per node.
// ---------------------------------------------------------------------------
__global__ __launch_bounds__(64) void init_quad_kernel(float* __restrict__ qx,
                                                       float* __restrict__ qw) {
  const int i = threadIdx.x;
  const int n = NQ;
  double x = cos(3.14159265358979323846 * ((double)i + 0.75) / ((double)n + 0.5));
  double p0 = 1.0, p1 = x;
  for (int it = 0; it < 64; ++it) {
    p0 = 1.0; p1 = x;
    for (int k = 2; k <= n; ++k) {
      double p2 = ((2.0 * k - 1.0) * x * p1 - (k - 1.0) * p0) / (double)k;
      p0 = p1; p1 = p2;
    }
    double dp = (double)n * (x * p1 - p0) / (x * x - 1.0);
    double dx = p1 / dp;
    x -= dx;
    if (fabs(dx) < 1e-15) break;
  }
  // derivative at converged root
  p0 = 1.0; p1 = x;
  for (int k = 2; k <= n; ++k) {
    double p2 = ((2.0 * k - 1.0) * x * p1 - (k - 1.0) * p0) / (double)k;
    p0 = p1; p1 = p2;
  }
  const double dp = (double)n * (x * p1 - p0) / (x * x - 1.0);
  const double w  = 2.0 / ((1.0 - x * x) * dp * dp);
  const double X  = ABND * x;
  qx[i] = (float)X;
  qw[i] = (float)(ABND * w * exp(-0.5 * X * X));
}

// ---------------------------------------------------------------------------
// Kernel 2: build g(delta) table with WMMA.
// One wave handles a tile of 16 deltas: G(16x1) = F(16x64) . w(64x1), done as
// two v_wmma_f32_16x16x32_f16 accumulating into the same f32 C (B-matrix has
// the weight vector replicated across all 16 columns, so every column of D
// holds the answer).
//
// f16 16x32 A layout (ISA 7.12.2): lane l holds row M = l&15; half = l>>4;
// element j<8  -> K = half*8 + j ; element j>=8 -> K = 16 + half*8 + (j-8).
// f32 16x16 C/D layout: lanes 0-15 VGPR j -> (M=j, N=lane); lanes 16-31
// VGPR j -> (M=j+8). So lane 0 owns g[base..base+7], lane 16 g[base+8..15],
// contiguous in its 8 accumulator VGPRs.
// ---------------------------------------------------------------------------
__global__ __launch_bounds__(256) void build_gtab_kernel(const float* __restrict__ qx,
                                                         const float* __restrict__ qw,
                                                         float* __restrict__ gtab) {
  const int lane = threadIdx.x & 31;
  const int half = lane >> 4;
  const int m    = lane & 15;
  const int tile = (blockIdx.x * blockDim.x + threadIdx.x) >> 5;
  const float dstep = DMAX / (float)(NTAB - 1);
  const float delta = (float)(tile * 16 + m) * dstep;

  v16h a0, b0, a1, b1;
#pragma unroll
  for (int j = 0; j < 16; ++j) {
    const int k = (j < 8) ? (half * 8 + j) : (16 + half * 8 + (j - 8));
    const float t0 = tanhf(delta * qx[k]);
    a0[j] = (_Float16)(1.0f - t0 * t0);
    b0[j] = (_Float16)qw[k];
    const float t1 = tanhf(delta * qx[32 + k]);
    a1[j] = (_Float16)(1.0f - t1 * t1);
    b1[j] = (_Float16)qw[32 + k];
  }

  v8f c = {};
  c = __builtin_amdgcn_wmma_f32_16x16x32_f16(false, a0, false, b0, (short)0, c, false, false);
  c = __builtin_amdgcn_wmma_f32_16x16x32_f16(false, a1, false, b1, (short)0, c, false, false);

  if (m == 0) {
    const int base = tile * 16 + half * 8;
#pragma unroll
    for (int j = 0; j < 8; ++j) gtab[base + j] = c[j] * INV_2PI;
  }
}

// ---------------------------------------------------------------------------
// Kernel 3: the serial scan. One wave; all 32 lanes redundantly carry the
// scalar recurrence (no broadcasts needed). u[] is double-buffered in LDS via
// CDNA5 async global->LDS loads; z[] chunks drain via async LDS->global
// stores; both synchronized with s_wait_asynccnt.
// ---------------------------------------------------------------------------
__global__ __launch_bounds__(32) void scan_kernel(const float* __restrict__ u,
                                                  const float* __restrict__ sig,
                                                  const float* __restrict__ gtab_g,
                                                  float* __restrict__ z,
                                                  int T) {
  __shared__ float s_gtab[NTAB + 1];
  __shared__ float s_u[2][CHUNK];
  __shared__ float s_z[2][CHUNK];

  const int lane = threadIdx.x;

  for (int i = lane; i < NTAB; i += 32) s_gtab[i] = gtab_g[i];
  if (lane == 0) {
    s_gtab[NTAB] = gtab_g[NTAB - 1];   // clamp entry for the lerp
    z[0] = 0.0f;                        // z_hist[0]
  }
  __builtin_amdgcn_s_barrier();

  const float sm1 = sig[0], sm2 = sig[1], sI = sig[2];
  const float sm1n1 = sig[3], sm2n2 = sig[4];
  const float sn1I = sig[5], sn2I = sig[6];
  const float sm1w = sig[7], sm2w = sig[8], sIw = sig[9];
  const float c1 = sm1 * sm1, c2 = sm2 * sm2, c3 = sI * sI;
  const float scale = (float)(NTAB - 1) / DMAX;

  float k1 = 0.0f, k2 = 0.0f, v = 0.0f;
  const int nchunks = (T + CHUNK - 1) / CHUNK;

  // Prefetch chunk 0 (async global -> LDS, 32 lanes x 16B = 512B = CHUNK f32).
  if (T >= CHUNK) {
    uint32_t ldsa = (uint32_t)(size_t)&s_u[0][lane * 4];
    uint64_t ga   = (uint64_t)(size_t)(u + lane * 4);
    asm volatile("global_load_async_to_lds_b128 %0, %1, off"
                 :: "v"(ldsa), "v"(ga) : "memory");
  } else {
    for (int i = lane; i < T; i += 32) s_u[0][i] = u[i];
  }

  for (int c = 0; c < nchunks; ++c) {
    const int bi = c & 1;

    // Wait: chunk-c load complete, chunk-(c-1) stores drained.
    asm volatile("s_wait_asynccnt 0" ::: "memory");

    // Prefetch chunk c+1 into the other buffer; overlaps the compute below.
    if (c + 1 < nchunks) {
      const int base = (c + 1) * CHUNK;
      if (base + CHUNK <= T) {
        uint32_t ldsa = (uint32_t)(size_t)&s_u[bi ^ 1][lane * 4];
        uint64_t ga   = (uint64_t)(size_t)(u + base + lane * 4);
        asm volatile("global_load_async_to_lds_b128 %0, %1, off"
                     :: "v"(ldsa), "v"(ga) : "memory");
      } else {
        for (int i = lane; i < T - base; i += 32) s_u[bi ^ 1][i] = u[base + i];
      }
    }

    int steps = T - c * CHUNK;
    if (steps > CHUNK) steps = CHUNK;

    // Serial recurrence: critical path = sqrt + LDS lerp + a few FMAs.
    for (int j = 0; j < steps; ++j) {
      const float in = s_u[bi][j];
      const float d2 = c1 * k1 * k1 + c2 * k2 * k2 + c3 * in * in;
      float f = sqrtf(d2) * scale;
      f = fminf(f, (float)(NTAB - 1));
      const int   i0 = (int)f;
      const float fr = f - (float)i0;
      const float g0 = s_gtab[i0];
      const float g  = g0 + fr * (s_gtab[i0 + 1] - g0);

      const float k1n = k1 + (-k1 + sm1n1 * g * k1 + sn1I * g * v) * RSTEP;
      const float k2n = k2 + (-k2 + sm2n2 * g * k2 + sn2I * g * v) * RSTEP;
      const float vn  = v  + (-v + in) * RSTEP;
      k1 = k1n; k2 = k2n; v = vn;

      const float zv = sm1w * g * k1 + sm2w * g * k2 + sIw * g * v;
      if (lane == 0) s_z[bi][j] = zv;
    }

    // Make the LDS writes visible, then drain z chunk asynchronously.
    asm volatile("s_wait_dscnt 0" ::: "memory");
#pragma unroll
    for (int q = 0; q < 4; ++q) {
      const int idx = q * 32 + lane;
      if (c * CHUNK + idx < T) {
        uint32_t ldsa = (uint32_t)(size_t)&s_z[bi][idx];
        uint64_t ga   = (uint64_t)(size_t)(z + 1 + c * CHUNK + idx);
        asm volatile("global_store_async_from_lds_b32 %0, %1, off"
                     :: "v"(ga), "v"(ldsa) : "memory");
      }
    }
  }
  asm volatile("s_wait_asynccnt 0" ::: "memory");
}

// ---------------------------------------------------------------------------
extern "C" void kernel_launch(void* const* d_in, const int* in_sizes, int n_in,
                              void* d_out, int out_size, void* d_ws, size_t ws_size,
                              hipStream_t stream) {
  const float* u   = (const float*)d_in[0];
  const float* sig = (const float*)d_in[1];
  float* out = (float*)d_out;
  const int T = in_sizes[0];

  float* qx   = (float*)d_ws;      // 64 floats
  float* qw   = qx + NQ;           // 64 floats
  float* gtab = qw + NQ;           // NTAB floats

  init_quad_kernel<<<1, 64, 0, stream>>>(qx, qw);
  // NTAB/16 = 256 tiles, one wave each; 8 waves per 256-thread block.
  build_gtab_kernel<<<(NTAB / 16) / 8, 256, 0, stream>>>(qx, qw, gtab);
  scan_kernel<<<1, 32, 0, stream>>>(u, sig, gtab, out, T);
}
